// NeuralFractionalSDE_1228360647362
// MI455X (gfx1250) — matrix operand order repaired and checked by reading, not compiled, not run on hardware
//
#include <hip/hip_runtime.h>
#include <hip/hip_bf16.h>

// ---------------------------------------------------------------------------
// NeuralFractionalSDE for MI455X (gfx1250, wave32, WMMA).
//
// Phase 1: fBm Hermitian IDFT (VALU, exact integer phase).
// Phase 2: Euler-Maruyama scan: per-step MLPs batched into
//          v_wmma_f32_16x16x32_f16 tiles, all weights LDS-resident,
//          branch-free v_tanh_f32 on the serial critical path.
// Phase 3: mean-path -> rank-1-collapsed conv -> flash attention -> hurst head.
// ---------------------------------------------------------------------------

typedef _Float16 half_t;
typedef _Float16 v16h __attribute__((ext_vector_type(16)));
typedef _Float16 v8h  __attribute__((ext_vector_type(8)));
typedef float    v8f  __attribute__((ext_vector_type(8)));

#define NB   32      // batch
#define NS   1024    // sequence
#define NP   64      // paths
#define NE   64      // embed
#define HALF 512

struct SdeW  { const float* w[16]; };  // dW0,db0,dW1,db1,dW2,db2,dW3,db3, then diff
struct ConvW { const float* w[4]; const float* bws[4]; };

// Branch-free tanh: hardware v_tanh_f32 when the builtin exists (gfx1250),
// else exp2/rcp sequence with no exec-mask divergence.
__device__ __forceinline__ float fast_tanh(float x) {
#if __has_builtin(__builtin_amdgcn_tanhf)
  return __builtin_amdgcn_tanhf(x);
#elif __has_builtin(__builtin_amdgcn_exp2f) && __has_builtin(__builtin_amdgcn_rcpf)
  float xc = fminf(fmaxf(x, -9.f), 9.f);
  float e2 = __builtin_amdgcn_exp2f(2.8853900817779268f * xc);   // e^(2x)
  return (e2 - 1.f) * __builtin_amdgcn_rcpf(e2 + 1.f);
#else
  float xc = fminf(fmaxf(x, -9.f), 9.f);
  float e2 = __expf(2.f * xc);
  return (e2 - 1.f) / (e2 + 1.f);
#endif
}

// ---------------- WMMA fragment helpers (contiguous k = 16h + s, same for A and B) -----

// A 16x32 f16: lanes 0-15 row M=lane, K = 0..15;
//              lanes 16-31 same rows,  K = 16..31.
__device__ __forceinline__ v16h load_a_f16(const half_t* A, int ld, int m0, int k0, int lane) {
  int m  = m0 + (lane & 15);
  int ko = k0 + ((lane >> 4) << 4);
  const half_t* p = A + m * ld + ko;
  v8h lo = *(const v8h*)p;
  v8h hi = *(const v8h*)(p + 8);
  return __builtin_shufflevector(lo, hi, 0,1,2,3,4,5,6,7,8,9,10,11,12,13,14,15);
}

// B 32x16 f16: lane = column N; lanes 0-15 hold K=k0..k0+15, lanes 16-31 K=k0+16..k0+31.
// Weights are stored TRANSPOSED in LDS (Wt[n][k], ld = K) so this is one contiguous 32B read.
__device__ __forceinline__ v16h load_b_f16(const half_t* Wt, int ld, int n0, int k0, int lane) {
  int n  = n0 + (lane & 15);
  int ko = k0 + ((lane >> 4) << 4);
  return *(const v16h*)(Wt + n * ld + ko);
}

__device__ __forceinline__ v8f wmma_f16(v16h a, v16h b, v8f c) {
  // (neg_a, A, neg_b, B, c_mod, C, reuse_a, reuse_b)
  return __builtin_amdgcn_wmma_f32_16x16x32_f16(false, a, false, b, (short)0, c, false, false);
}

// ---------------- Phase 1: fBm via exact-phase Hermitian IDFT --------------

__global__ void k_fbm(const float* __restrict__ zre, const float* __restrict__ zim,
                      float* __restrict__ fbm) {
  __shared__ float za[HALF], zb[HALF];
  const int bp  = blockIdx.x;          // 0..2047 = b*64+p
  const int tid = threadIdx.x;
  const float PI = 3.14159265358979f;
  for (int j = tid; j < HALF; j += 256) {
    float amp = sqrtf(0.5f * __powf(2.f * __sinf(PI * (float)(j + 1) / (float)NS), 1.4f));
    za[j] = zre[bp * HALF + j] * amp;
    zb[j] = zim[bp * HALF + j] * amp;
  }
  __syncthreads();
  const float scale = __powf(1.f / (float)NS, 0.7f) / (float)NS;  // ifft 1/S * dt0^H
  for (int n = tid; n < NS; n += 256) {
    float acc = 0.f;
    for (int j = 0; j < HALF - 1; ++j) {
      int ph = ((j + 1) * n) & (NS - 1);          // exact: theta = 2*pi*ph/NS
      float s, c;
      __sincosf((float)ph * (2.f * PI / (float)NS), &s, &c);
      acc += za[j] * c - zb[j] * s;
    }
    float nyq = (n & 1) ? -za[HALF - 1] : za[HALF - 1];
    fbm[(size_t)bp * NS + n] = (2.f * acc + nyq) * scale;
  }
}

// ---------------- Phase 2: Euler-Maruyama scan with WMMA MLPs --------------
// One block = one batch element (its 64 paths). 128 threads = 4 waves.
// LDS carve (bytes):
#define L_WT1 0        // [2][128][64] f16 : 32768
#define L_WT2 32768    // [2][64][128] f16 : 32768
#define L_H1  65536    // [64][64]  f16 : 8192
#define L_H2  73728    // [64][128] f16 : 16384
#define L_H3  90112    // [64][64]  f16 : 8192
#define L_B1  98304    // [2][128] f32 : 1024
#define L_B2  99328    // [2][64]  f32 : 512
#define L_W0  99840    // [2][128] f32 : 1024
#define L_B0  100864   // [2][64]  f32 : 512
#define L_W3  101376   // [2][64]  f32 : 512
#define L_B3  101888   // [2] f32 (padded 32)
#define L_Y   101920   // [64] f32
#define L_F   102176   // [64] f32
#define L_G   102432   // [64] f32
#define SCAN_LDS_BYTES 102688

__global__ __launch_bounds__(128) void k_sde_scan(const float* __restrict__ t,
                                                  const float* __restrict__ y0,
                                                  const float* __restrict__ fbm,
                                                  SdeW W, float* __restrict__ paths) {
  extern __shared__ char smem[];
  half_t* wt1  = (half_t*)(smem + L_WT1);
  half_t* wt2  = (half_t*)(smem + L_WT2);
  half_t* h1   = (half_t*)(smem + L_H1);
  half_t* h2   = (half_t*)(smem + L_H2);
  half_t* h3   = (half_t*)(smem + L_H3);
  float*  b1s  = (float*)(smem + L_B1);
  float*  b2s  = (float*)(smem + L_B2);
  float*  W0s  = (float*)(smem + L_W0);
  float*  b0s  = (float*)(smem + L_B0);
  float*  W3s  = (float*)(smem + L_W3);
  float*  b3s  = (float*)(smem + L_B3);
  float*  ybuf = (float*)(smem + L_Y);
  float*  fbuf = (float*)(smem + L_F);
  float*  gbuf = (float*)(smem + L_G);

  const int tid = threadIdx.x;
  const int b   = blockIdx.x;           // batch index; rows = b*64 + p, p = 0..63

  // --- preload weights into LDS (transposed, f16) once for the whole scan ---
  for (int m = 0; m < 2; ++m) {
    const float* W1 = W.w[m * 8 + 2];   // [64][128]
    const float* W2 = W.w[m * 8 + 4];   // [128][64]
    for (int idx = tid; idx < 8192; idx += 128) {
      int n = idx >> 6, k = idx & 63;
      wt1[m * 8192 + idx] = (half_t)W1[k * 128 + n];
    }
    for (int idx = tid; idx < 8192; idx += 128) {
      int n = idx >> 7, k = idx & 127;
      wt2[m * 8192 + idx] = (half_t)W2[k * 64 + n];
    }
    if (tid < 128) { b1s[m * 128 + tid] = W.w[m * 8 + 3][tid];
                     W0s[m * 128 + tid] = W.w[m * 8 + 0][tid]; }
    if (tid < 64)  { b2s[m * 64 + tid]  = W.w[m * 8 + 5][tid];
                     b0s[m * 64 + tid]  = W.w[m * 8 + 1][tid];
                     W3s[m * 64 + tid]  = W.w[m * 8 + 6][tid]; }
    if (tid == 0)  b3s[m] = W.w[m * 8 + 7][0];
  }
  if (tid < 64) {
    float y = y0[b];                    // D=1, same y0 for all paths of b
    ybuf[tid] = y;
    paths[((size_t)b * NP + tid) * NS] = y;
  }

  const int lane = tid & 31;
  const int m0   = (tid >> 5) << 4;     // wave's 16-row M tile
  const int col  = lane & 15;
  const int rsel = lane >> 4;

  for (int s = 0; s < NS - 1; ++s) {
    float tval = t[b * NS + s];
    float dtv  = t[b * NS + s + 1] - tval;
    __syncthreads();                    // ybuf (and on s==0: weights) ready

    for (int m = 0; m < 2; ++m) {       // m=0 drift, m=1 diffusion
      // layer0 (K=2) on VALU: h1 = tanh(t*W0[0] + y*W0[1] + b0)
      for (int idx = tid; idx < 64 * 64; idx += 128) {
        int r = idx >> 6, j = idx & 63;
        float z = tval * W0s[m * 128 + j] + ybuf[r] * W0s[m * 128 + 64 + j] + b0s[m * 64 + j];
        h1[idx] = (half_t)fast_tanh(z);
      }
      __syncthreads();

      // layer1: [64x64] x [64x128] -> h2 (WMMA, K tiles 0/32, 8 N tiles)
      {
        v16h a0 = load_a_f16(h1, 64, m0, 0, lane);
        v16h a1 = load_a_f16(h1, 64, m0, 32, lane);
        const half_t* wB = wt1 + m * 8192;
        #pragma unroll
        for (int nt = 0; nt < 8; ++nt) {
          int n0 = nt << 4;
          float bias = b1s[m * 128 + n0 + col];
          v8f c;
          #pragma unroll
          for (int i = 0; i < 8; ++i) c[i] = bias;
          c = wmma_f16(a0, load_b_f16(wB, 64, n0, 0, lane), c);
          c = wmma_f16(a1, load_b_f16(wB, 64, n0, 32, lane), c);
          #pragma unroll
          for (int i = 0; i < 8; ++i) {
            int mm = m0 + i + (rsel << 3);           // C/D layout: VGPR i -> M=i / M=8+i
            h2[mm * 128 + n0 + col] = (half_t)fast_tanh(c[i]);
          }
        }
      }
      __syncthreads();

      // layer2: [64x128] x [128x64] -> h3 (WMMA, 4 K tiles, 4 N tiles)
      {
        v16h a[4];
        #pragma unroll
        for (int kt = 0; kt < 4; ++kt) a[kt] = load_a_f16(h2, 128, m0, kt * 32, lane);
        const half_t* wB = wt2 + m * 8192;
        #pragma unroll
        for (int nt = 0; nt < 4; ++nt) {
          int n0 = nt << 4;
          float bias = b2s[m * 64 + n0 + col];
          v8f c;
          #pragma unroll
          for (int i = 0; i < 8; ++i) c[i] = bias;
          #pragma unroll
          for (int kt = 0; kt < 4; ++kt)
            c = wmma_f16(a[kt], load_b_f16(wB, 128, n0, kt * 32, lane), c);
          #pragma unroll
          for (int i = 0; i < 8; ++i) {
            int mm = m0 + i + (rsel << 3);
            h3[mm * 64 + n0 + col] = (half_t)fast_tanh(c[i]);
          }
        }
      }
      __syncthreads();

      // layer3 (N=1) on VALU: dot(h3[row], W3) + b3 ; diffusion -> softplus
      if (tid < 64) {
        float acc = b3s[m];
        const half_t* hr = h3 + tid * 64;
        #pragma unroll
        for (int jv = 0; jv < 8; ++jv) {
          v8h hv = *(const v8h*)(hr + jv * 8);       // ds_load_b128
          #pragma unroll
          for (int u2 = 0; u2 < 8; ++u2)
            acc += (float)hv[u2] * W3s[m * 64 + jv * 8 + u2];
        }
        if (m == 0) fbuf[tid] = acc;
        else        gbuf[tid] = (acc > 20.f) ? acc : log1pf(__expf(acc));
      }
      __syncthreads();
    }

    // Euler-Maruyama update; dB streamed from fbm with prefetch-ahead.
    if (tid < 64) {
      size_t row = (size_t)b * NP + tid;
      float f0 = fbm[row * NS + s];
      float f1 = fbm[row * NS + s + 1];
      __builtin_prefetch(&fbm[row * NS + s + 16], 0, 1);   // global_prefetch_b8
      float y = ybuf[tid] + fbuf[tid] * dtv + gbuf[tid] * (f1 - f0);
      ybuf[tid] = y;
      paths[row * NS + s + 1] = y;
    }
  }
}

// ---------------- Phase 3a: mean over paths --------------------------------

__global__ void k_meanpath(const float* __restrict__ paths, float* __restrict__ meanp) {
  int idx = blockIdx.x * 256 + threadIdx.x;      // B*S = 32768
  if (idx >= NB * NS) return;
  int b = idx >> 10, s = idx & (NS - 1);
  const float* p = paths + (size_t)(b * NP) * NS + s;
  float acc = 0.f;
  for (int pp = 0; pp < NP; ++pp) acc += p[(size_t)pp * NS];
  meanp[idx] = acc * (1.f / (float)NP);
}

// ---------------- Phase 3b: rank-1 conv collapse u/v tables ----------------
// feat[b,s,:] = mean[b,s]*projW + projb  (rank-1 in channels) so
// conv_i contribution = mean[b,s'] * (projW . convW_i[tau,:,e]) + (projb . convW_i[tau,:,e])

__global__ void k_uv(const float* __restrict__ projW, const float* __restrict__ projb,
                     ConvW cw, float* __restrict__ u, float* __restrict__ v) {
  int idx = blockIdx.x * 256 + threadIdx.x;      // 4*11*64 = 2816
  if (idx >= 4 * 11 * 64) return;
  int i = idx / (11 * 64);
  int rem = idx % (11 * 64);
  int tau = rem >> 6, e = rem & 63;
  const int KS[4] = {3, 5, 7, 11};
  float uu = 0.f, vv = 0.f;
  if (tau < KS[i]) {
    const float* w = cw.w[i] + (size_t)(tau * NE) * NE + e;  // convW_i[tau][c][e]
    for (int c = 0; c < NE; ++c) {
      float wc = w[(size_t)c * NE];
      uu += projW[c] * wc;
      vv += projb[c] * wc;
    }
  }
  u[idx] = uu;
  v[idx] = vv;
}

// ---------------- Phase 3c: conv (26 taps/out) -> x f16 --------------------

__global__ void k_conv(const float* __restrict__ meanp, const float* __restrict__ u,
                       const float* __restrict__ v, ConvW cw, half_t* __restrict__ x) {
  int idx = blockIdx.x * 256 + threadIdx.x;      // B*S*E = 2097152 exactly
  int e = idx & 63, s = (idx >> 6) & (NS - 1), b = idx >> 16;
  const int KS[4] = {3, 5, 7, 11};
  float acc = 0.f;
  #pragma unroll
  for (int i = 0; i < 4; ++i) {
    int off = KS[i] >> 1;
    for (int tau = 0; tau < KS[i]; ++tau) {
      int sp = s + tau - off;
      if (sp >= 0 && sp < NS) {
        int ui = (i * 11 + tau) * 64 + e;
        acc += meanp[b * NS + sp] * u[ui] + v[ui];
      }
    }
    acc += cw.bws[i][e];
  }
  x[idx] = (half_t)(acc * 0.25f);
}

// ---------------- Phase 3d: flash attention per (b, head) ------------------

#define ATTN_LDS_BYTES (NS * 16 * 2 * 2 + 64)    // K + V f16 + osum

__global__ void k_attn(const half_t* __restrict__ x,
                       const float* __restrict__ Wq, const float* __restrict__ Wk,
                       const float* __restrict__ Wv,
                       const float* __restrict__ bq, const float* __restrict__ bk,
                       const float* __restrict__ bv, float* __restrict__ omean) {
  extern __shared__ char smem[];
  half_t* kb   = (half_t*)smem;                  // [1024][16]
  half_t* vb   = (half_t*)(smem + NS * 16 * 2);  // [1024][16]
  float*  osum = (float*)(smem + NS * 16 * 4);
  const int tid = threadIdx.x;
  const int b = blockIdx.x >> 2, h = blockIdx.x & 3;
  const int hoff = h * 16;
  if (tid < 16) osum[tid] = 0.f;

  for (int idx = tid; idx < NS * 16; idx += 256) {
    int s = idx >> 4, d = idx & 15;
    float ak = bk[hoff + d], av = bv[hoff + d];
    const half_t* xr = x + (size_t)(b * NS + s) * NE;
    for (int c = 0; c < NE; ++c) {
      float xf = (float)xr[c];
      ak += xf * Wk[c * NE + hoff + d];
      av += xf * Wv[c * NE + hoff + d];
    }
    kb[idx] = (half_t)ak;
    vb[idx] = (half_t)av;
  }
  __syncthreads();

  for (int sq = tid; sq < NS; sq += 256) {
    float q[16];
    #pragma unroll
    for (int d = 0; d < 16; ++d) q[d] = bq[hoff + d];
    const half_t* xr = x + (size_t)(b * NS + sq) * NE;
    for (int c = 0; c < NE; ++c) {
      float xf = (float)xr[c];
      #pragma unroll
      for (int d = 0; d < 16; ++d) q[d] += xf * Wq[c * NE + hoff + d];
    }
    // online softmax over 1024 keys
    float mx = -3.4e38f, l = 0.f;
    float acc[16];
    #pragma unroll
    for (int d = 0; d < 16; ++d) acc[d] = 0.f;
    for (int ks = 0; ks < NS; ++ks) {
      const half_t* kr = kb + ks * 16;
      float sc = 0.f;
      #pragma unroll
      for (int d = 0; d < 16; ++d) sc += q[d] * (float)kr[d];
      sc *= 0.25f;                                 // 1/sqrt(16)
      float nm = fmaxf(mx, sc);
      float es = __expf(mx - nm);
      float ew = __expf(sc - nm);
      l = l * es + ew;
      const half_t* vr = vb + ks * 16;
      #pragma unroll
      for (int d = 0; d < 16; ++d) acc[d] = acc[d] * es + ew * (float)vr[d];
      mx = nm;
    }
    float inv = 1.f / l;
    #pragma unroll
    for (int d = 0; d < 16; ++d) atomicAdd(&osum[d], acc[d] * inv);  // ds_add_f32
  }
  __syncthreads();
  if (tid < 16) omean[b * NE + hoff + tid] = osum[tid] * (1.f / (float)NS);
}

// ---------------- Phase 3e: Wo + hurst head --------------------------------

__global__ void k_head(const float* __restrict__ omean, const float* __restrict__ Wo,
                       const float* __restrict__ bo,
                       const float* __restrict__ hW0, const float* __restrict__ hb0,
                       const float* __restrict__ hW1, const float* __restrict__ hb1,
                       float* __restrict__ hurst) {
  __shared__ float featv[64], hh[64];
  int b = blockIdx.x, e = threadIdx.x;
  float f = bo[e];
  for (int c = 0; c < 64; ++c) f += omean[b * 64 + c] * Wo[c * 64 + e];
  featv[e] = f;
  __syncthreads();
  float a = hb0[e];
  for (int c = 0; c < 64; ++c) a += featv[c] * hW0[c * 64 + e];
  hh[e] = fmaxf(a, 0.f);
  __syncthreads();
  if (e == 0) {
    float sacc = hb1[0];
    for (int c = 0; c < 64; ++c) sacc += hh[c] * hW1[c];
    hurst[b] = 1.f / (1.f + __expf(-sacc));
  }
}

// ---------------- host launch ----------------------------------------------
// d_in order (setup_inputs dict, params flattened in insertion order):
//  0:t 1:y0 2:z_re 3:z_im
//  4..11  drift W0,b0,W1,b1,W2,b2,W3,b3
// 12..19  diff  W0,b0,W1,b1,W2,b2,W3,b3
// 20:proj_W 21:proj_b
// 22..29  conv W0,b0,W1,b1,W2,b2,W3,b3
// 30:Wq 31:Wk 32:Wv 33:Wo 34:bq 35:bk 36:bv 37:bo
// 38:hurst_W0 39:hurst_b0 40:hurst_W1 41:hurst_b1 42:n_paths(int)

extern "C" void kernel_launch(void* const* d_in, const int* in_sizes, int n_in,
                              void* d_out, int out_size, void* d_ws, size_t ws_size,
                              hipStream_t stream) {
  (void)in_sizes; (void)n_in; (void)out_size; (void)ws_size;
  auto fp = [&](int i) { return (const float*)d_in[i]; };

  const float* t   = fp(0);
  const float* y0  = fp(1);
  const float* zre = fp(2);
  const float* zim = fp(3);
  SdeW sw;
  for (int i = 0; i < 16; ++i) sw.w[i] = fp(4 + i);
  const float* projW = fp(20); const float* projb = fp(21);
  ConvW cw;
  for (int i = 0; i < 4; ++i) { cw.w[i] = fp(22 + 2 * i); cw.bws[i] = fp(23 + 2 * i); }
  const float* Wq = fp(30); const float* Wk = fp(31);
  const float* Wv = fp(32); const float* Wo = fp(33);
  const float* bq = fp(34); const float* bk = fp(35);
  const float* bv = fp(36); const float* bo = fp(37);
  const float* hW0 = fp(38); const float* hb0 = fp(39);
  const float* hW1 = fp(40); const float* hb1 = fp(41);

  float* paths = (float*)d_out;                              // [B,P,S,1]
  float* hurst = paths + (size_t)NB * NP * NS;               // [B,1]

  // workspace carve (~12.8 MB)
  char* ws = (char*)d_ws;
  float*  fbm   = (float*)(ws + 0);               // [2048][1024] f32 : 8388608 B
  float*  meanp = (float*)(ws + 8388608);         // [32][1024]  f32 : 131072 B
  float*  u     = (float*)(ws + 8519680);         // [4][11][64] f32 : 11264 B
  float*  v     = (float*)(ws + 8530944);         // [4][11][64] f32 : 11264 B
  half_t* x     = (half_t*)(ws + 8542208);        // [32][1024][64] f16 : 4194304 B
  float*  omean = (float*)(ws + 12736512);        // [32][64] f32 : 8192 B

  k_fbm     <<<NB * NP, 256, 0, stream>>>(zre, zim, fbm);
  k_sde_scan<<<NB, 128, SCAN_LDS_BYTES, stream>>>(t, y0, fbm, sw, paths);
  k_meanpath<<<(NB * NS) / 256, 256, 0, stream>>>(paths, meanp);
  k_uv      <<<11, 256, 0, stream>>>(projW, projb, cw, u, v);
  k_conv    <<<(NB * NS * NE) / 256, 256, 0, stream>>>(meanp, u, v, cw, x);
  k_attn    <<<NB * 4, 256, ATTN_LDS_BYTES, stream>>>(x, Wq, Wk, Wv, bq, bk, bv, omean);
  k_head    <<<NB, 64, 0, stream>>>(omean, Wo, bo, hW0, hb0, hW1, hb1, hurst);
}
